// Head_27444841022172
// MI455X (gfx1250) — compile-verified, matrix operand order checked by reading, not run on hardware
//
#include <hip/hip_runtime.h>

// ---------------------------------------------------------------------------
// Fused causal single-head attention for MI455X (gfx1250, wave32, WMMA + TDM).
//   B=8, T=2048, E=1024, H=64
// Kernel 0: Wq/Wk/Wv f32 -> bf16, transposed [m][n][k]   (TDM-copyable form)
// Kernel 1: QKV projection; W K-slices DMA'd by tensor_load_to_lds (double
//           buffered), x A-tiles as b128 loads + v_cvt_pk_bf16_f32.
//           Q/K stored row-major bf16 (scale folded into Q), V transposed.
// Kernel 2: flash attention; K/V tiles DMA'd per wave by tensor_load_to_lds
//           (double buffered, TENSORcnt-pipelined), WMMA QK^T + online
//           softmax + WMMA P.V, P transposed C->A via LDS bounce.
// ---------------------------------------------------------------------------

#define B_   8
#define T_   2048
#define E_   1024
#define H_   64
#define MTOT (B_ * T_)          // 16384 total rows

typedef __attribute__((ext_vector_type(16))) __bf16 v16bf;
typedef __attribute__((ext_vector_type(16))) short  v16s;
typedef __attribute__((ext_vector_type(8)))  short  v8s;   // one b128 chunk
typedef __attribute__((ext_vector_type(8)))  float  v8f;
typedef __attribute__((ext_vector_type(4)))  float  v4f;
typedef __attribute__((ext_vector_type(4)))  unsigned int u32x4;
typedef __attribute__((ext_vector_type(8)))  unsigned int u32x8;

union Tile {
  v16s s;
  v16bf bf;
  v8s  h[2];     // h[0] = K chunk kb..kb+7, h[1] = kb+16..kb+23 (A/B layout)
};

// f32 -> bf16 via native convert
__device__ __forceinline__ unsigned short f2bf(float f) {
  __bf16 h = (__bf16)f;
  return __builtin_bit_cast(unsigned short, h);
}

__device__ __forceinline__ v8f wmma_bf16(const Tile& a, const Tile& b, v8f c) {
  // 8-arg form: (neg_a, A, neg_b, B, c_mod, C, reuse_a, reuse_b)
  return __builtin_amdgcn_wmma_f32_16x16x32_bf16(false, a.bf, false, b.bf,
                                                 (short)0, c, false, false);
}

// ---------------------------------------------------------------------------
// TDM: build a D# (groups 0+1, ISA 8.3/8.4) and issue tensor_load_to_lds.
// 2D tile: tile_d0 x tile_d1 elements of 2 bytes, row stride in elements.
// All inputs must be wave-uniform (descriptor lives in SGPRs).
// ---------------------------------------------------------------------------
__device__ __forceinline__ void tdm_load_2d(unsigned lds_off,
                                            unsigned long long gaddr,
                                            unsigned tile_d0, unsigned tile_d1,
                                            unsigned tensor_d0, unsigned tensor_d1,
                                            unsigned long long row_stride_elems) {
  u32x4 g0;
  g0[0] = 1u;                                          // count=1, user descriptor
  g0[1] = lds_off;                                     // lds_addr (bytes)
  g0[2] = (unsigned)gaddr;                             // global_addr[31:0]
  g0[3] = (unsigned)((gaddr >> 32) & 0x1FFFFFFull)     // global_addr[56:32]
        | (2u << 30);                                  // type=2 ("image")
  u32x8 g1;
  g1[0] = 1u << 16;                                    // data_size=1 (2B); mask=0
  g1[1] = (tensor_d0 & 0xFFFFu) << 16;                 // tensor_dim0[15:0] @63:48
  g1[2] = (tensor_d0 >> 16) | ((tensor_d1 & 0xFFFFu) << 16); // dim0 hi | dim1 lo
  g1[3] = ((tensor_d1 >> 16) & 0xFFFFu)                // dim1 hi @111:96
        | ((tile_d0 & 0xFFFFu) << 16);                 // tile_dim0 @127:112
  g1[4] = tile_d1 & 0xFFFFu;                           // tile_dim1 (0 => 1D)
  g1[5] = (unsigned)row_stride_elems;                  // tensor_dim0_stride lo
  g1[6] = (unsigned)((row_stride_elems >> 32) & 0xFFFFull); // stride hi
  g1[7] = 0u;                                          // dim1_stride (unused, 2D)
  asm volatile("tensor_load_to_lds %0, %1" :: "s"(g0), "s"(g1) : "memory");
}

// ---------------------------------------------------------------------------
// Kernel 0: W prep -- f32 [E][H] -> bf16 transposed [m][n(64)][k(1024)]
// ---------------------------------------------------------------------------
__global__ __launch_bounds__(256)
void wprep_kernel(const float* __restrict__ Wq, const float* __restrict__ Wk,
                  const float* __restrict__ Wv, unsigned short* __restrict__ Wt) {
  const int idx = blockIdx.x * 256 + threadIdx.x;      // 0 .. 3*H*E-1
  const int m   = idx / (H_ * E_);
  const int rem = idx - m * (H_ * E_);
  const int n   = rem / E_;
  const int k   = rem - n * E_;
  const float* W = (m == 0) ? Wq : ((m == 1) ? Wk : Wv);
  Wt[idx] = f2bf(W[k * H_ + n]);
}

// ---------------------------------------------------------------------------
// Kernel 1: Q/K/V projection.  4 waves/block, each wave computes a 16x64 tile
// of Q, K and V (A tile of x loaded once -> 12 WMMAs per 32-K step).
// W K-slices (3 x 64x32 bf16 = 12 KB) are TDM-DMA'd into double-buffered LDS:
// wave m (m<3) issues the descriptor for matrix m, TENSORcnt pipelines the
// next slice against the current slice's WMMAs.
// ---------------------------------------------------------------------------
__global__ __launch_bounds__(128)
void qkv_proj_kernel(const float* __restrict__ x,
                     const unsigned short* __restrict__ Wt,  // [3][64][1024] bf16
                     unsigned short* __restrict__ Qb,
                     unsigned short* __restrict__ Kb,
                     unsigned short* __restrict__ Vt) {
  __shared__ __align__(16) unsigned short Wl[2][3][32 * H_];  // 24 KB double buffer

  const int lane = threadIdx.x & 31;
  const int wavu = __builtin_amdgcn_readfirstlane((int)(threadIdx.x >> 5));
  const int half = lane >> 4;                 // A/B layout lane-half K-chunk
  const int l16  = lane & 15;
  const int r0   = (blockIdx.x * 4 + wavu) * 16;   // 16 output rows per wave

  const unsigned wl_base = (unsigned)(size_t)&Wl[0][0][0];
  const unsigned long long wt_base = (unsigned long long)(size_t)Wt;

  v8f accQ[4], accK[4], accV[4];
  #pragma unroll
  for (int nt = 0; nt < 4; ++nt) { accQ[nt] = {}; accK[nt] = {}; accV[nt] = {}; }

  const int NIT = E_ / 32;                    // 32 K-steps

  // prologue: DMA slice 0 into buffer 0 (one matrix per wave, waves 0..2)
  if (wavu < 3) {
    tdm_load_2d(wl_base + (unsigned)((0 * 3 + wavu) * 32 * H_) * 2u,
                wt_base + (unsigned long long)(wavu * H_ * E_ + 0) * 2ull,
                /*tile*/ 32u, H_, /*tensor*/ 32u, H_, /*stride*/ E_);
  }

  for (int it = 0; it < NIT; ++it) {
    const int kk  = it * 32;
    const int buf = it & 1;

    __syncthreads();   // everyone done reading buf^1 (iteration it-1)
    if (it + 1 < NIT) {
      if (wavu < 3) {
        tdm_load_2d(wl_base + (unsigned)(((1 - buf) * 3 + wavu) * 32 * H_) * 2u,
                    wt_base + (unsigned long long)(wavu * H_ * E_ + kk + 32) * 2ull,
                    32u, H_, 32u, H_, E_);
      }
      __builtin_amdgcn_s_wait_tensorcnt(1);   // slice `it` landed (in-order)
    } else {
      __builtin_amdgcn_s_wait_tensorcnt(0);
    }
    __syncthreads();   // slice `it` visible to all waves

    // ---- A tile of x: 16x32 bf16 in WMMA A-layout (b128 loads + cvt) ----
    Tile a;
    {
      const int row = r0 + l16;
      const float* xp = x + (long)row * E_ + kk + half * 8;
      v4f x0 = *(const v4f*)(xp);
      v4f x1 = *(const v4f*)(xp + 4);
      v4f x2 = *(const v4f*)(xp + 16);
      v4f x3 = *(const v4f*)(xp + 20);
      #pragma unroll
      for (int i = 0; i < 4; ++i) {
        a.bf[i]      = (__bf16)x0[i];
        a.bf[4 + i]  = (__bf16)x1[i];
        a.bf[8 + i]  = (__bf16)x2[i];
        a.bf[12 + i] = (__bf16)x3[i];
      }
    }

    // ---- B tiles from LDS (2 x ds_load_b128 each); 12 WMMAs / K-step ----
    const unsigned short* wq = &Wl[buf][0][0];
    const unsigned short* wk = &Wl[buf][1][0];
    const unsigned short* wv = &Wl[buf][2][0];
    #pragma unroll
    for (int nt = 0; nt < 4; ++nt) {
      const int col = (nt * 16 + l16) * 32 + half * 8;
      Tile bq, bk, bv;
      bq.h[0] = *(const v8s*)&wq[col];
      bq.h[1] = *(const v8s*)&wq[col + 16];
      bk.h[0] = *(const v8s*)&wk[col];
      bk.h[1] = *(const v8s*)&wk[col + 16];
      bv.h[0] = *(const v8s*)&wv[col];
      bv.h[1] = *(const v8s*)&wv[col + 16];
      accQ[nt] = wmma_bf16(a, bq, accQ[nt]);
      accK[nt] = wmma_bf16(a, bk, accK[nt]);
      accV[nt] = wmma_bf16(a, bv, accV[nt]);
    }
  }

  // ---- store C tiles as bf16 ----
  // Q (scale 1/sqrt(64) folded) and K row-major [row][h]; V transposed to
  // [b][h][t] so the attention kernel's V tiles are TDM-friendly.
  const int b_    = r0 / T_;                  // 16-row tile never straddles a batch
  const int jbase = r0 - b_ * T_;
  #pragma unroll
  for (int nt = 0; nt < 4; ++nt) {
    #pragma unroll
    for (int g = 0; g < 8; ++g) {
      const int row = r0 + g + 8 * half;      // C layout: VGPR g, lane-half -> row
      const int col = nt * 16 + l16;
      Qb[row * H_ + col] = f2bf(accQ[nt][g] * 0.125f);
      Kb[row * H_ + col] = f2bf(accK[nt][g]);
      Vt[((long)b_ * H_ + col) * T_ + jbase + g + 8 * half] = f2bf(accV[nt][g]);
    }
  }
}

// ---------------------------------------------------------------------------
// Kernel 2: flash attention.  One wave per 16-query tile, 4 waves/block.
// Per wave, TDM double-buffers the 32-key K tile (contiguous 4 KB) and V tile
// (64 feature-rows x 32 keys from transposed Vt) into LDS while WMMAs consume
// the previous tile.  Per 32-key step: 4 WMMAs S=QK^T, online softmax
// (shfl_xor within lane halves), P C->A transpose via LDS, 4 WMMAs O += P.V.
// ---------------------------------------------------------------------------
__global__ __launch_bounds__(128)
void attn_kernel(const unsigned short* __restrict__ Qb,
                 const unsigned short* __restrict__ Kb,
                 const unsigned short* __restrict__ Vt,
                 const long long* __restrict__ lengths,
                 float* __restrict__ out) {
  __shared__ __align__(16) unsigned short Kl[4][2][32 * H_];  // 32 KB
  __shared__ __align__(16) unsigned short Vl[4][2][H_ * 32];  // 32 KB
  __shared__ __align__(16) unsigned short Pl[4][16 * 32];     //  4 KB

  const int lane = threadIdx.x & 31;
  const int wavu = __builtin_amdgcn_readfirstlane((int)(threadIdx.x >> 5));
  const int half = lane >> 4;
  const int l16  = lane & 15;

  const int qtile = blockIdx.x * 4 + wavu;    // 0..1023
  const int b     = qtile >> 7;               // 128 q-tiles per batch
  const int q0    = (qtile & 127) * 16;
  const int len   = (int)lengths[b];

  const unsigned short* Qp = Qb + (long)(b * T_ + q0) * H_;
  const unsigned long long kb_base =
      (unsigned long long)(size_t)(Kb + (long)(b * T_) * H_);
  const unsigned long long vt_base =
      (unsigned long long)(size_t)(Vt + (long)b * H_ * T_);
  const unsigned kl_off = (unsigned)(size_t)&Kl[wavu][0][0];
  const unsigned vl_off = (unsigned)(size_t)&Vl[wavu][0][0];

  // Q A-tiles over head dim 64 (two 16x32 chunks), loaded once (b128 loads)
  Tile aQ[2];
  #pragma unroll
  for (int t = 0; t < 2; ++t) {
    const unsigned short* qp = Qp + l16 * H_ + t * 32 + half * 8;
    aQ[t].h[0] = *(const v8s*)(qp);
    aQ[t].h[1] = *(const v8s*)(qp + 16);
  }

  float m8[8], l8[8];
  v8f accO[4];
  #pragma unroll
  for (int g = 0; g < 8; ++g) { m8[g] = -3.0e38f; l8[g] = 0.f; }
  #pragma unroll
  for (int nt = 0; nt < 4; ++nt) accO[nt] = {};

  const int nkeys = q0 + 16;                  // causal: keys [0, q0+16)
  const int NIT   = (nkeys + 31) >> 5;

  // prologue: DMA K/V tiles for j0=0 into buffer 0
  tdm_load_2d(kl_off, kb_base, /*tile*/ 32u * H_, 0u,
              /*tensor*/ 32u * H_, 1u, /*stride*/ 32u * H_);           // K: 1D 4KB
  tdm_load_2d(vl_off, vt_base, 32u, H_, 32u, H_, (unsigned long long)T_); // V: 2D

  for (int it = 0; it < NIT; ++it) {
    const int j0  = it << 5;
    const int buf = it & 1;

    if (it + 1 < NIT) {   // DMA next tiles into the other buffer, then wait
      const unsigned long long koff = (unsigned long long)((j0 + 32) * H_) * 2ull;
      tdm_load_2d(kl_off + (1u - buf) * (32u * H_ * 2u), kb_base + koff,
                  32u * H_, 0u, 32u * H_, 1u, 32u * H_);
      tdm_load_2d(vl_off + (1u - buf) * (32u * H_ * 2u),
                  vt_base + (unsigned long long)(j0 + 32) * 2ull,
                  32u, H_, 32u, H_, (unsigned long long)T_);
      __builtin_amdgcn_s_wait_tensorcnt(2);   // current pair landed (in-order)
    } else {
      __builtin_amdgcn_s_wait_tensorcnt(0);
    }
    asm volatile("" ::: "memory");

    const unsigned short* kbuf = &Kl[wavu][buf][0];   // [key][feat]
    const unsigned short* vbuf = &Vl[wavu][buf][0];   // [feat][key]

    // ---- S = Q K^T for 32 keys (two 16x16 C tiles) ----
    v8f sc[2];
    #pragma unroll
    for (int c = 0; c < 2; ++c) {
      const int jj = c * 16 + l16;            // key (local) owned by this lane
      const unsigned short* kp = kbuf + jj * H_;
      Tile bK0, bK1;
      bK0.h[0] = *(const v8s*)(kp + half * 8);
      bK0.h[1] = *(const v8s*)(kp + half * 8 + 16);
      bK1.h[0] = *(const v8s*)(kp + 32 + half * 8);
      bK1.h[1] = *(const v8s*)(kp + 32 + half * 8 + 16);
      v8f s0 = {};
      s0 = wmma_bf16(aQ[0], bK0, s0);
      s0 = wmma_bf16(aQ[1], bK1, s0);
      // causal + length mask (scale folded into Q); -3e38 (not -inf) keeps the
      // online-softmax alpha NaN-free; the diagonal is always in range so
      // fully-masked partial tiles are annihilated later by alpha.
      const int j = j0 + jj;
      #pragma unroll
      for (int g = 0; g < 8; ++g) {
        const int q  = q0 + g + 8 * half;
        const bool ok = (j <= q) && (((j < len) && (q < len)) || (j == q));
        if (!ok) s0[g] = -3.0e38f;
      }
      sc[c] = s0;
    }

    // ---- online softmax; write P (bf16) to this wave's LDS region ----
    #pragma unroll
    for (int g = 0; g < 8; ++g) {
      float t = fmaxf(sc[0][g], sc[1][g]);
      #pragma unroll
      for (int msk = 1; msk < 16; msk <<= 1)      // reduce within 16-lane half
        t = fmaxf(t, __shfl_xor(t, msk, 32));
      const float mnew  = fmaxf(m8[g], t);
      const float alpha = __expf(m8[g] - mnew);
      m8[g] = mnew;
      const float p0 = __expf(sc[0][g] - mnew);
      const float p1 = __expf(sc[1][g] - mnew);
      float rs = p0 + p1;
      #pragma unroll
      for (int msk = 1; msk < 16; msk <<= 1)
        rs += __shfl_xor(rs, msk, 32);
      l8[g] = l8[g] * alpha + rs;
      #pragma unroll
      for (int nt = 0; nt < 4; ++nt) accO[nt][g] *= alpha;
      const int row = g + 8 * half;
      Pl[wavu][row * 32 + l16]      = f2bf(p0);
      Pl[wavu][row * 32 + 16 + l16] = f2bf(p1);
    }

    // Wave-local LDS RAW: DS ops are in-order per wave; asm is a compiler
    // fence + explicit s_wait_dscnt per the CDNA5 counter model.
    asm volatile("s_wait_dscnt 0" ::: "memory");

    // ---- read P back in A layout (16x32): two ds_load_b128 per lane ----
    Tile aP;
    {
      const unsigned short* pp = &Pl[wavu][l16 * 32 + half * 8];
      aP.h[0] = *(const v8s*)(pp);
      aP.h[1] = *(const v8s*)(pp + 16);
    }

    // ---- O += P . V  (V B-tiles from LDS [feat][key]: b128 reads) ----
    #pragma unroll
    for (int nt = 0; nt < 4; ++nt) {
      const int n = nt * 16 + l16;
      const unsigned short* vp = vbuf + n * 32 + half * 8;
      Tile bV;
      bV.h[0] = *(const v8s*)(vp);
      bV.h[1] = *(const v8s*)(vp + 16);
      accO[nt] = wmma_bf16(aP, bV, accO[nt]);
    }
  }

  // ---- normalize rows and store f32 output ----
  #pragma unroll
  for (int nt = 0; nt < 4; ++nt) {
    #pragma unroll
    for (int g = 0; g < 8; ++g) {
      const int row = q0 + g + 8 * half;
      const int col = nt * 16 + l16;
      out[((long)b * T_ + row) * H_ + col] = accO[nt][g] / l8[g];
    }
  }
}

// ---------------------------------------------------------------------------
// Host launch
// ---------------------------------------------------------------------------
extern "C" void kernel_launch(void* const* d_in, const int* in_sizes, int n_in,
                              void* d_out, int out_size, void* d_ws, size_t ws_size,
                              hipStream_t stream) {
  const float*     x       = (const float*)d_in[0];
  const float*     Wq      = (const float*)d_in[1];
  const float*     Wk      = (const float*)d_in[2];
  const float*     Wv      = (const float*)d_in[3];
  const long long* lengths = (const long long*)d_in[4];   // int64 lengths
  float*           out     = (float*)d_out;

  // Workspace: Q/K row-major bf16, V transposed bf16 (2 MB each), W bf16^T
  unsigned short* Qb = (unsigned short*)d_ws;
  unsigned short* Kb = Qb + (size_t)MTOT * H_;
  unsigned short* Vt = Kb + (size_t)MTOT * H_;
  unsigned short* Wt = Vt + (size_t)MTOT * H_;            // 3*64*1024 bf16

  wprep_kernel<<<(3 * H_ * E_) / 256, 256, 0, stream>>>(Wq, Wk, Wv, Wt);
  qkv_proj_kernel<<<MTOT / 16 / 4, 128, 0, stream>>>(x, Wt, Qb, Kb, Vt);
  attn_kernel<<<(MTOT / 16) / 4, 128, 0, stream>>>(Qb, Kb, Vt, lengths, out);
}
